// CirculantLinear_v3_11785390260448
// MI455X (gfx1250) — matrix operand order repaired and checked by reading, not compiled, not run on hardware
//
#include <hip/hip_runtime.h>
#include <hip/hip_bf16.h>

// Block-circulant linear == dense GEMM  C[32768x1024] = X @ W^T with
// W[o][i] = eigens[o>>3][i>>3][(o-i)&7].
//
// Fast path (ws >= 132MB): pre-split X and W into bf16 hi/lo arrays once,
// then a pure-WMMA GEMM loop (split-precision: ahi*bhi + alo*bhi + ahi*blo).
// Fallback (small ws): previous kernel that splits X in-register per tile.

typedef __attribute__((ext_vector_type(16))) __bf16 v16bf;
typedef __attribute__((ext_vector_type(8)))  __bf16 v8bf;
typedef __attribute__((ext_vector_type(8)))  float  v8f;

#define IN_CH  1024
#define OUT_CH 1024
#define MINI   8
#define GXB    128
#define BATCH  32768

__device__ __forceinline__ unsigned short f2bf_u(float f) {
  unsigned u = __builtin_bit_cast(unsigned, f);
  unsigned r = u + 0x7FFFu + ((u >> 16) & 1u);   // round-to-nearest-even
  return (unsigned short)(r >> 16);
}
__device__ __forceinline__ float bfu2f(unsigned short h) {
  unsigned u = ((unsigned)h) << 16;
  return __builtin_bit_cast(float, u);
}

// ---------------- Kernel 1: expand eigens into split-bf16 weight matrices ---
__global__ __launch_bounds__(256) void expand_w_kernel(
    const float* __restrict__ eigens,
    unsigned short* __restrict__ whi,
    unsigned short* __restrict__ wlo) {
  int idx = blockIdx.x * 256 + threadIdx.x;     // idx = o*1024 + i
  int o = idx >> 10;
  int i = idx & 1023;
  float w = eigens[(((o >> 3) * GXB) + (i >> 3)) * MINI + ((o - i) & 7)];
  unsigned short h = f2bf_u(w);
  float rem = w - bfu2f(h);
  whi[idx] = h;
  wlo[idx] = f2bf_u(rem);
}

// ---------------- Kernel 2: split X into bf16 hi/lo (one-time, amortized) ---
__global__ __launch_bounds__(256) void split_x_kernel(
    const float* __restrict__ X,
    unsigned short* __restrict__ xhi,
    unsigned short* __restrict__ xlo) {
  size_t base = ((size_t)blockIdx.x * 256 + threadIdx.x) * 4;
  float4 f = *(const float4*)(X + base);
  float fv[4] = {f.x, f.y, f.z, f.w};
  unsigned short hv[4], lv[4];
  #pragma unroll
  for (int e = 0; e < 4; ++e) {
    hv[e] = f2bf_u(fv[e]);
    lv[e] = f2bf_u(fv[e] - bfu2f(hv[e]));
  }
  *(ushort4*)(xhi + base) = make_ushort4(hv[0], hv[1], hv[2], hv[3]);
  *(ushort4*)(xlo + base) = make_ushort4(lv[0], lv[1], lv[2], lv[3]);
}

// Build a 16x32 bf16 A fragment for this lane from pre-split bf16 X.
// elem idx 0..15 -> K = kk + half*16 + idx
__device__ __forceinline__ v16bf load_a_frag(const unsigned short* Xs,
                                             size_t rowbase, int half) {
  v8bf c0 = *(const v8bf*)(Xs + rowbase + half * 16);
  v8bf c1 = *(const v8bf*)(Xs + rowbase + half * 16 + 8);
  return __builtin_shufflevector(c0, c1, 0, 1, 2, 3, 4, 5, 6, 7,
                                 8, 9, 10, 11, 12, 13, 14, 15);
}

// ---------------- Kernel 3a: WMMA GEMM, pre-split X (fast path) -------------
// Block: 256 threads = 8 waves. Wave tile 32x32 (2x2 WMMA tiles).
// Block tile: M=128 (4 waves) x N=64 (2 waves). Grid: (N/64, BATCH/128).
__global__ __launch_bounds__(256) void bcirc_gemm_presplit(
    const unsigned short* __restrict__ Xhi,
    const unsigned short* __restrict__ Xlo,
    const unsigned short* __restrict__ Whi,
    const unsigned short* __restrict__ Wlo,
    float* __restrict__ Out) {
  const int lane = threadIdx.x & 31;
  const int wave = threadIdx.x >> 5;
  const int half = lane >> 4;
  const int l15  = lane & 15;

  const int m0 = blockIdx.y * 128 + (wave >> 1) * 32;
  const int n0 = blockIdx.x * 64  + (wave & 1)  * 32;

  const size_t arow0 = (size_t)(m0 + l15) * IN_CH;
  const size_t arow1 = (size_t)(m0 + 16 + l15) * IN_CH;
  const size_t brow0 = (size_t)(n0 + l15) * IN_CH + half * 16;
  const size_t brow1 = (size_t)(n0 + 16 + l15) * IN_CH + half * 16;

  v8f acc[2][2] = {};

  #pragma unroll 2
  for (int kk = 0; kk < IN_CH; kk += 32) {
    v16bf ahi[2], alo[2], bhi[2], blo[2];
    ahi[0] = load_a_frag(Xhi, arow0 + kk, half);
    ahi[1] = load_a_frag(Xhi, arow1 + kk, half);
    alo[0] = load_a_frag(Xlo, arow0 + kk, half);
    alo[1] = load_a_frag(Xlo, arow1 + kk, half);
    bhi[0] = *(const v16bf*)(Whi + brow0 + kk);
    bhi[1] = *(const v16bf*)(Whi + brow1 + kk);
    blo[0] = *(const v16bf*)(Wlo + brow0 + kk);
    blo[1] = *(const v16bf*)(Wlo + brow1 + kk);

    #pragma unroll
    for (int mi = 0; mi < 2; ++mi) {
      #pragma unroll
      for (int ni = 0; ni < 2; ++ni) {
        acc[mi][ni] = __builtin_amdgcn_wmma_f32_16x16x32_bf16(
            false, ahi[mi], false, bhi[ni], (short)0, acc[mi][ni], false, false);
        acc[mi][ni] = __builtin_amdgcn_wmma_f32_16x16x32_bf16(
            false, alo[mi], false, bhi[ni], (short)0, acc[mi][ni], false, false);
        acc[mi][ni] = __builtin_amdgcn_wmma_f32_16x16x32_bf16(
            false, ahi[mi], false, blo[ni], (short)0, acc[mi][ni], false, false);
      }
    }
  }

  #pragma unroll
  for (int mi = 0; mi < 2; ++mi) {
    #pragma unroll
    for (int ni = 0; ni < 2; ++ni) {
      int col = n0 + ni * 16 + l15;
      #pragma unroll
      for (int r = 0; r < 8; ++r) {
        int row = m0 + mi * 16 + half * 8 + r;
        Out[(size_t)row * OUT_CH + col] = acc[mi][ni][r];
      }
    }
  }
}

// ---------------- Kernel 3b: WMMA GEMM, in-register X split (fallback) ------
__global__ __launch_bounds__(256) void bcirc_gemm_kernel(
    const float* __restrict__ X,
    const unsigned short* __restrict__ Whi,
    const unsigned short* __restrict__ Wlo,
    float* __restrict__ Out) {
  const int lane = threadIdx.x & 31;
  const int wave = threadIdx.x >> 5;
  const int half = lane >> 4;
  const int l15  = lane & 15;

  const int m0 = blockIdx.y * 128 + (wave >> 1) * 32;
  const int n0 = blockIdx.x * 64  + (wave & 1)  * 32;

  v8f acc[2][2] = {};

  #pragma unroll 1
  for (int kk = 0; kk < IN_CH; kk += 32) {
    v16bf ahi[2], alo[2];
    #pragma unroll
    for (int mi = 0; mi < 2; ++mi) {
      const float* xp = X + (size_t)(m0 + mi * 16 + l15) * IN_CH + kk + half * 16;
      float4 c0a = *(const float4*)(xp);
      float4 c0b = *(const float4*)(xp + 4);
      float4 c1a = *(const float4*)(xp + 8);
      float4 c1b = *(const float4*)(xp + 12);
      float fv[16] = {c0a.x, c0a.y, c0a.z, c0a.w, c0b.x, c0b.y, c0b.z, c0b.w,
                      c1a.x, c1a.y, c1a.z, c1a.w, c1b.x, c1b.y, c1b.z, c1b.w};
      #pragma unroll
      for (int e = 0; e < 16; ++e) {
        unsigned short h = f2bf_u(fv[e]);
        float rem = fv[e] - bfu2f(h);
        ahi[mi][e] = __builtin_bit_cast(__bf16, h);
        alo[mi][e] = __builtin_bit_cast(__bf16, f2bf_u(rem));
      }
    }
    v16bf bhi[2], blo[2];
    #pragma unroll
    for (int ni = 0; ni < 2; ++ni) {
      size_t off = (size_t)(n0 + ni * 16 + l15) * IN_CH + kk + half * 16;
      bhi[ni] = *(const v16bf*)(Whi + off);
      blo[ni] = *(const v16bf*)(Wlo + off);
    }
    #pragma unroll
    for (int mi = 0; mi < 2; ++mi) {
      #pragma unroll
      for (int ni = 0; ni < 2; ++ni) {
        acc[mi][ni] = __builtin_amdgcn_wmma_f32_16x16x32_bf16(
            false, ahi[mi], false, bhi[ni], (short)0, acc[mi][ni], false, false);
        acc[mi][ni] = __builtin_amdgcn_wmma_f32_16x16x32_bf16(
            false, alo[mi], false, bhi[ni], (short)0, acc[mi][ni], false, false);
        acc[mi][ni] = __builtin_amdgcn_wmma_f32_16x16x32_bf16(
            false, ahi[mi], false, blo[ni], (short)0, acc[mi][ni], false, false);
      }
    }
  }

  #pragma unroll
  for (int mi = 0; mi < 2; ++mi) {
    #pragma unroll
    for (int ni = 0; ni < 2; ++ni) {
      int col = n0 + ni * 16 + l15;
      #pragma unroll
      for (int r = 0; r < 8; ++r) {
        int row = m0 + mi * 16 + half * 8 + r;
        Out[(size_t)row * OUT_CH + col] = acc[mi][ni][r];
      }
    }
  }
}

extern "C" void kernel_launch(void* const* d_in, const int* in_sizes, int n_in,
                              void* d_out, int out_size, void* d_ws, size_t ws_size,
                              hipStream_t stream) {
  const float* x      = (const float*)d_in[0];   // [32768, 1024] fp32
  const float* eigens = (const float*)d_in[1];   // [128, 128, 8] fp32
  float* out = (float*)d_out;                    // [32768, 1024] fp32

  const size_t wElems = (size_t)OUT_CH * IN_CH;    // 1M
  const size_t xElems = (size_t)BATCH * IN_CH;     // 32M

  unsigned short* whi = (unsigned short*)d_ws;
  unsigned short* wlo = whi + wElems;

  expand_w_kernel<<<(int)(wElems / 256), 256, 0, stream>>>(eigens, whi, wlo);

  dim3 grid(OUT_CH / 64, BATCH / 128);
  const size_t need = (2 * wElems + 2 * xElems) * sizeof(unsigned short);
  if (ws_size >= need) {
    // Fast path: convert X once, then pure-WMMA GEMM.
    unsigned short* xhi = wlo + wElems;
    unsigned short* xlo = xhi + xElems;
    split_x_kernel<<<(int)(xElems / 4 / 256), 256, 0, stream>>>(x, xhi, xlo);
    bcirc_gemm_presplit<<<grid, 256, 0, stream>>>(xhi, xlo, whi, wlo, out);
  } else {
    // Fallback: split X in-register inside the GEMM loop.
    bcirc_gemm_kernel<<<grid, 256, 0, stream>>>(x, whi, wlo, out);
  }
}
